// MultiHeadAttention_27891517620512
// MI455X (gfx1250) — matrix operand order repaired and checked by reading, not compiled, not run on hardware
//
#include <hip/hip_runtime.h>
#include <hip/hip_bf16.h>

// MHA forward for B=2,S=2048,D=1024,H=16,DK=64 on gfx1250 (wave32, WMMA bf16).
// d_out = [output (B*S*D f32)] ++ [alpha (B*H*S*S f32)].

typedef __attribute__((ext_vector_type(16))) __bf16 v16bf;
typedef __attribute__((ext_vector_type(8)))  __bf16 v8bf;
typedef __attribute__((ext_vector_type(8)))  float  v8f;
typedef __attribute__((ext_vector_type(4)))  float  v4f;

#define CB  2
#define CS  2048
#define CD  1024
#define CH  16
#define CDK 64
#define NEG_BIG (-3.4028234663852886e38f)   // jnp.finfo(f32).min, matches reference masking

__device__ __forceinline__ v8f wmma_bf16(v16bf a, v16bf b, v8f c) {
    return __builtin_amdgcn_wmma_f32_16x16x32_bf16(false, a, false, b, (short)0, c, false, false);
}

union vfrag { v16bf v; v8bf h[2]; };

// ---------------------------------------------------------------- fp32 -> bf16 (vectorized)
__global__ void f2bf_kernel(const float* __restrict__ in, __bf16* __restrict__ out, int n4) {
    int i = blockIdx.x * blockDim.x + threadIdx.x;
    int stride = gridDim.x * blockDim.x;
    for (; i < n4; i += stride) {
        v4f x = ((const v4f*)in)[i];
        __bf16 y[4] = { (__bf16)x.x, (__bf16)x.y, (__bf16)x.z, (__bf16)x.w };
        ((v8bf*)0, 0);  // no-op
        *(uint64_t*)(out + (size_t)i * 4) = *(const uint64_t*)y;
    }
}

// ---------------------------------------------------------------- projection GEMM
// out[row, col] = sum_k A[row,k] * W[col,k] + bias[col]   (A: 4096 x 1024, W: 1024 x 1024)
// Per block (4 waves): 128 rows x 64 cols.  Per wave: 32 rows x 64 cols
// (two A row-halves share each B fragment -> 8 WMMA per K32 chunk).
// mode 0: bf16 out in [B,H,S,DK]            (Q, K)
// mode 1: bf16 out in [B,H,DK,S] transposed (V)
// mode 2: f32  out row-major [4096,1024]    (final output)
__global__ __launch_bounds__(128) void proj_gemm_kernel(
        const __bf16* __restrict__ A, const __bf16* __restrict__ W,
        const float* __restrict__ bias, void* __restrict__ out, int mode) {
    const int wave = threadIdx.x >> 5;
    const int lane = threadIdx.x & 31;
    const int grp  = lane >> 4;
    const int ln   = lane & 15;
    const int kb   = grp * 8;

    const int mrow = blockIdx.x * 128 + wave * 32;  // token tile base (32 rows per wave)
    const int ncol = blockIdx.y * 64;               // feature tile base

    const __bf16* Arow0 = A + (size_t)(mrow + ln) * CD;
    const __bf16* Arow1 = Arow0 + (size_t)16 * CD;

    v8f acc0[4], acc1[4];
#pragma unroll
    for (int nt = 0; nt < 4; ++nt) { acc0[nt] = (v8f){}; acc1[nt] = (v8f){}; }

    for (int k0 = 0; k0 < CD; k0 += 32) {
        vfrag a0, a1;
        a0.h[0] = *(const v8bf*)(Arow0 + k0 + 2 * kb);
        a0.h[1] = *(const v8bf*)(Arow0 + k0 + 2 * kb + 8);
        a1.h[0] = *(const v8bf*)(Arow1 + k0 + 2 * kb);
        a1.h[1] = *(const v8bf*)(Arow1 + k0 + 2 * kb + 8);
#pragma unroll
        for (int nt = 0; nt < 4; ++nt) {
            v16bf wb = *(const v16bf*)(W + (size_t)(ncol + nt * 16 + ln) * CD + k0 + grp * 16);
            acc0[nt] = wmma_bf16(a0.v, wb, acc0[nt]);
            acc1[nt] = wmma_bf16(a1.v, wb, acc1[nt]);
        }
    }

#pragma unroll
    for (int half = 0; half < 2; ++half) {
#pragma unroll
        for (int nt = 0; nt < 4; ++nt) {
#pragma unroll
            for (int r = 0; r < 8; ++r) {
                float av = half ? acc1[nt][r] : acc0[nt][r];
                int row = mrow + half * 16 + r + 8 * grp;  // token index in [0, B*S)
                int col = ncol + nt * 16 + ln;             // feature in [0, D)
                float v = av + bias[col];
                if (mode == 2) {
                    ((float*)out)[(size_t)row * CD + col] = v;
                } else {
                    int b = row >> 11, s = row & (CS - 1);
                    int h = col >> 6,  dk = col & (CDK - 1);
                    __bf16* o = (__bf16*)out;
                    if (mode == 0)
                        o[(((size_t)(b * CH + h)) * CS + s) * CDK + dk] = (__bf16)v;  // [B,H,S,DK]
                    else
                        o[(((size_t)(b * CH + h)) * CDK + dk) * CS + s] = (__bf16)v;  // [B,H,DK,S]
                }
            }
        }
    }
}

// ---------------------------------------------------------------- fused attention
// grid.x = B*H, grid.y = S/64; block = 128 (4 waves); each wave owns 16 query rows.
// Pass 1: per-lane online softmax stats (no per-tile shuffles), one butterfly merge.
// Pass 2: recompute scores, single normalized alpha write, P->A-frag transpose via
//         wave-private padded LDS, context accumulate with WMMA.
__global__ __launch_bounds__(128) void attn_kernel(
        const __bf16* __restrict__ Q, const __bf16* __restrict__ K,
        const __bf16* __restrict__ Vt, const int* __restrict__ pad,
        const int* __restrict__ causal_ptr,
        float* __restrict__ alpha, __bf16* __restrict__ ctx) {
    __shared__ __bf16 lds_p[4][16 * 40];             // 80B row stride: conflict-free b128 reads

    const int wave = threadIdx.x >> 5;
    const int lane = threadIdx.x & 31;
    const int grp  = lane >> 4;
    const int ln   = lane & 15;
    const int kb   = grp * 8;

    const int bh    = blockIdx.x;
    const int b     = bh / CH;
    const int h     = bh % CH;
    const int qtile = blockIdx.y * 4 + wave;
    const int qbase = qtile * 16;
    const int causal = causal_ptr[0];
    const float scl = 0.125f;                        // 1/sqrt(DK)

    const __bf16* Qh = Q  + (size_t)bh * CS * CDK;
    const __bf16* Kh = K  + (size_t)bh * CS * CDK;
    const __bf16* Vh = Vt + (size_t)bh * CDK * CS;
    float* alph = alpha + (size_t)bh * CS * CS + (size_t)qbase * CS;

    // Q A-fragments for both K32 chunks of DK=64 (held for the whole kernel)
    v16bf qa[2];
#pragma unroll
    for (int kc = 0; kc < 2; ++kc) {
        vfrag f;
        const __bf16* p = Qh + (size_t)(qbase + ln) * CDK + kc * 32 + 2 * kb;
        f.h[0] = *(const v8bf*)p;
        f.h[1] = *(const v8bf*)(p + 8);
        qa[kc] = f.v;
    }

    const int jmax = causal ? qtile : (CS / 16 - 1);

    // ---------------- pass 1: per-lane online max/sum ----------------
    float mL[8], lL[8];
#pragma unroll
    for (int r = 0; r < 8; ++r) { mL[r] = -INFINITY; lL[r] = 0.0f; }

    for (int j = 0; j <= jmax; ++j) {
        const __bf16* kp = Kh + (size_t)(j * 16 + ln) * CDK + grp * 16;
        if (j + 4 <= jmax)
            __builtin_prefetch(Kh + (size_t)((j + 4) * 16 + ln) * CDK + grp * 16, 0, 0);
        v16bf kb0 = *(const v16bf*)kp;
        v16bf kb1 = *(const v16bf*)(kp + 32);
        v8f c = (v8f){};
        c = wmma_bf16(qa[0], kb0, c);
        c = wmma_bf16(qa[1], kb1, c);
        int colg = j * 16 + ln;
        bool pm = pad[b * CS + colg] != 0;
#pragma unroll
        for (int r = 0; r < 8; ++r) {
            float s = c[r] * scl;
            int rowg = qbase + r + 8 * grp;
            if (pm || (causal && colg > rowg)) s = NEG_BIG;
            float Mn = fmaxf(mL[r], s);
            lL[r] = lL[r] * __expf(mL[r] - Mn) + __expf(s - Mn);
            mL[r] = Mn;
        }
    }

    // single cross-lane butterfly merge (16-lane groups)
    float M[8], Linv[8];
#pragma unroll
    for (int r = 0; r < 8; ++r) {
        float mx = mL[r];
        mx = fmaxf(mx, __shfl_xor(mx, 1, 32));
        mx = fmaxf(mx, __shfl_xor(mx, 2, 32));
        mx = fmaxf(mx, __shfl_xor(mx, 4, 32));
        mx = fmaxf(mx, __shfl_xor(mx, 8, 32));
        float l = lL[r] * __expf(mL[r] - mx);
        l += __shfl_xor(l, 1, 32);
        l += __shfl_xor(l, 2, 32);
        l += __shfl_xor(l, 4, 32);
        l += __shfl_xor(l, 8, 32);
        M[r] = mx;
        Linv[r] = 1.0f / l;
    }

    // ---------------- pass 2: alpha write (single pass) + context ----------------
    v8f accv[4];
#pragma unroll
    for (int nt = 0; nt < 4; ++nt) accv[nt] = (v8f){};

    for (int jp = 0; jp < CS / 32; ++jp) {
        const int j0 = jp * 2;
        const bool live = (j0 <= jmax);              // wave-uniform
        if (live) {
            if (j0 + 4 <= jmax)
                __builtin_prefetch(Kh + (size_t)((j0 + 4) * 16 + ln) * CDK + grp * 16, 0, 0);
            __builtin_prefetch(Vh + (size_t)(ln * 2) * CS + (jp + 1) * 32, 0, 0);
        }
#pragma unroll
        for (int t = 0; t < 2; ++t) {
            int j = j0 + t;
            float pv[8];
            if (j <= jmax) {
                const __bf16* kp = Kh + (size_t)(j * 16 + ln) * CDK + grp * 16;
                v16bf kb0 = *(const v16bf*)kp;
                v16bf kb1 = *(const v16bf*)(kp + 32);
                v8f c = (v8f){};
                c = wmma_bf16(qa[0], kb0, c);
                c = wmma_bf16(qa[1], kb1, c);
                int colg = j * 16 + ln;
                bool pm = pad[b * CS + colg] != 0;
#pragma unroll
                for (int r = 0; r < 8; ++r) {
                    float s = c[r] * scl;
                    int rowg = qbase + r + 8 * grp;
                    if (pm || (causal && colg > rowg)) s = NEG_BIG;
                    pv[r] = __expf(s - M[r]) * Linv[r];
                }
            } else {
#pragma unroll
                for (int r = 0; r < 8; ++r) pv[r] = 0.0f;
            }
#pragma unroll
            for (int r = 0; r < 8; ++r) {
                alph[(size_t)(r + 8 * grp) * CS + j * 16 + ln] = pv[r];   // only alpha write
                lds_p[wave][(r + 8 * grp) * 40 + t * 16 + ln] = (__bf16)pv[r];
            }
        }
        if (live) {
            // transpose P (C-layout) -> A-fragment via wave-private LDS
            vfrag pf;
            pf.h[0] = *(const v8bf*)&lds_p[wave][ln * 40 + 2 * kb];
            pf.h[1] = *(const v8bf*)&lds_p[wave][ln * 40 + 2 * kb + 8];
#pragma unroll
            for (int nt = 0; nt < 4; ++nt) {
                v16bf vb = *(const v16bf*)(Vh + (size_t)(nt * 16 + ln) * CS + jp * 32 + grp * 16);
                accv[nt] = wmma_bf16(pf.v, vb, accv[nt]);
            }
        }
    }

    // context tile -> bf16 [B,S,D] (heads concatenated along D)
    __bf16* cbase = ctx + ((size_t)b * CS + qbase) * CD + h * CDK;
#pragma unroll
    for (int nt = 0; nt < 4; ++nt)
#pragma unroll
        for (int r = 0; r < 8; ++r)
            cbase[(size_t)(r + 8 * grp) * CD + nt * 16 + ln] = (__bf16)accv[nt][r];
}

// ---------------------------------------------------------------- launcher
extern "C" void kernel_launch(void* const* d_in, const int* in_sizes, int n_in,
                              void* d_out, int out_size, void* d_ws, size_t ws_size,
                              hipStream_t stream) {
    const float* X  = (const float*)d_in[0];
    const float* Wq = (const float*)d_in[1];
    const float* bq = (const float*)d_in[2];
    const float* Wk = (const float*)d_in[3];
    const float* bk = (const float*)d_in[4];
    const float* Wv = (const float*)d_in[5];
    const float* bv = (const float*)d_in[6];
    const float* Wo = (const float*)d_in[7];
    const float* bo = (const float*)d_in[8];
    const int* pad    = (const int*)d_in[9];
    const int* causal = (const int*)d_in[10];

    float* out   = (float*)d_out;                 // [B,S,D]
    float* alpha = out + (size_t)CB * CS * CD;    // [B,H,S,S]

    // workspace carve-up (bf16), ~50 MB total
    __bf16* w = (__bf16*)d_ws;
    const size_t nX = (size_t)CB * CS * CD;       // 4M
    const size_t nW = (size_t)CD * CD;            // 1M
    __bf16* Xb  = w;            w += nX;
    __bf16* Wqb = w;            w += nW;
    __bf16* Wkb = w;            w += nW;
    __bf16* Wvb = w;            w += nW;
    __bf16* Wob = w;            w += nW;
    __bf16* Qb  = w;            w += nX;          // [B,H,S,DK]
    __bf16* Kb  = w;            w += nX;          // [B,H,S,DK]
    __bf16* Vtb = w;            w += nX;          // [B,H,DK,S]
    __bf16* Ctx = w;            w += nX;          // [B,S,D]

    // 1) convert inputs to bf16
    f2bf_kernel<<<1024, 256, 0, stream>>>(X,  Xb,  (int)(nX / 4));
    f2bf_kernel<<<1024, 256, 0, stream>>>(Wq, Wqb, (int)(nW / 4));
    f2bf_kernel<<<1024, 256, 0, stream>>>(Wk, Wkb, (int)(nW / 4));
    f2bf_kernel<<<1024, 256, 0, stream>>>(Wv, Wvb, (int)(nW / 4));
    f2bf_kernel<<<1024, 256, 0, stream>>>(Wo, Wob, (int)(nW / 4));

    // 2) QKV projections (WMMA bf16 GEMM, head-layout epilogues)
    dim3 pgrid(CB * CS / 128, CD / 64);
    proj_gemm_kernel<<<pgrid, 128, 0, stream>>>(Xb, Wqb, bq, (void*)Qb,  0);
    proj_gemm_kernel<<<pgrid, 128, 0, stream>>>(Xb, Wkb, bk, (void*)Kb,  0);
    proj_gemm_kernel<<<pgrid, 128, 0, stream>>>(Xb, Wvb, bv, (void*)Vtb, 1);

    // 3) fused scores + softmax + alpha + context
    attn_kernel<<<dim3(CB * CH, CS / 64), 128, 0, stream>>>(Qb, Kb, Vtb, pad, causal,
                                                            alpha, Ctx);

    // 4) output projection (f32 epilogue straight into d_out)
    proj_gemm_kernel<<<pgrid, 128, 0, stream>>>(Ctx, Wob, bo, (void*)out, 2);
}
